// GAT_degree_38989713113446
// MI455X (gfx1250) — matrix operand order repaired and checked by reading, not compiled, not run on hardware
//
#include <hip/hip_runtime.h>
#include <cstdint>

typedef __attribute__((ext_vector_type(16))) _Float16 v16h;
typedef __attribute__((ext_vector_type(8)))  float    v8f;
typedef long long i64;

union V16H { v16h v; _Float16 h[16]; uint32_t u[8]; uint4 q[2]; };

#define NN   4096
#define FF   512
#define HH   4
#define DD1  256
#define OO1  64
#define NEGC (-9e15f)

__device__ __forceinline__ float leakyf(float x) { return x >= 0.f ? x : 0.2f * x; }
__device__ __forceinline__ float eluf(float x)   { return x > 0.f ? x : (__expf(x) - 1.f); }

// ---------------------------------------------------------------------------
// h_in = x + (obs==1) * theta  -> f16
// ---------------------------------------------------------------------------
__global__ void prep_hin_kernel(const float* __restrict__ x, const int* __restrict__ obs,
                                const float* __restrict__ theta, _Float16* __restrict__ hin) {
  i64 idx = (i64)blockIdx.x * blockDim.x + threadIdx.x;
  if (idx >= (i64)NN * FF) return;
  int i = (int)(idx / FF), f = (int)(idx % FF);
  float v = x[idx];
  if (obs[i] == 1) v += theta[f];
  hin[idx] = (_Float16)v;
}

// ---------------------------------------------------------------------------
// Transpose+convert weight: W[K,N] f32 -> WT[N,K] f16 (batched)
// ---------------------------------------------------------------------------
__global__ void cvtT_kernel(const float* __restrict__ W, i64 sW,
                            _Float16* __restrict__ WT, i64 sWT, int K, int N) {
  W += (i64)blockIdx.z * sW; WT += (i64)blockIdx.z * sWT;
  int idx = blockIdx.x * blockDim.x + threadIdx.x;
  if (idx >= K * N) return;
  int k = idx / N, n = idx % N;
  WT[(i64)n * K + k] = (_Float16)W[idx];
}

// ---------------------------------------------------------------------------
// WMMA GEMM: C[M,N](f32) = A[M,K](f16,row) @ B, with B given as BT[N,K](f16,row).
// Optionally also writes CT[N,M] f16 (for use as a WMMA B-operand downstream).
// One wave per 16x16 tile; K-loop in steps of 32 via v_wmma_f32_16x16x32_f16.
// A frag: lanes 0-15 = rows, K=hh*16+0..15 contiguous
// -> two 16B loads per lane.
// B frag from BT rows: lane = column n, K=hh*16+0..15 contiguous -> 32B.
// ---------------------------------------------------------------------------
__global__ __launch_bounds__(128) void wmma_gemm_kernel(
    const _Float16* __restrict__ A, i64 sA,
    const _Float16* __restrict__ BT, i64 sBT,
    float* __restrict__ C, i64 sC,
    _Float16* __restrict__ CT, i64 sCT,
    int M, int N, int K) {
  int batch = blockIdx.z;
  A += batch * sA; BT += batch * sBT;
  if (C)  C  += batch * sC;
  if (CT) CT += batch * sCT;
  int lane = threadIdx.x & 31, wave = threadIdx.x >> 5;
  int ntile = blockIdx.x * 4 + wave;
  int mtile = blockIdx.y;
  if (ntile * 16 >= N) return;
  int hh = lane >> 4;
  int mrow = mtile * 16 + (lane & 15);
  int ncol = ntile * 16 + (lane & 15);
  const _Float16* arow = A + (i64)mrow * K;
  const _Float16* brow = BT + (i64)ncol * K;
  v8f acc = {};
  for (int kb = 0; kb < K; kb += 32) {
    V16H a, b;
    a.q[0] = *(const uint4*)(arow + kb + hh * 16);
    a.q[1] = *(const uint4*)(arow + kb + hh * 16 + 8);
    b.q[0] = *(const uint4*)(brow + kb + hh * 16);
    b.q[1] = *(const uint4*)(brow + kb + hh * 16 + 8);
    acc = __builtin_amdgcn_wmma_f32_16x16x32_f16(false, a.v, false, b.v,
                                                 (short)0, acc, false, false);
  }
#pragma unroll
  for (int r = 0; r < 8; r++) {
    int m = mtile * 16 + hh * 8 + r;
    float vv = acc[r];
    if (C)  C[(i64)m * N + ncol] = vv;
    if (CT) CT[(i64)ncol * M + m] = (_Float16)vv;
  }
}

// ---------------------------------------------------------------------------
// f1 = Wh @ a[:D], f2 = Wh @ a[D:]   (wave per row)
// ---------------------------------------------------------------------------
__global__ __launch_bounds__(256) void coef_kernel(
    const float* __restrict__ Wh, i64 sWh, int D,
    const float* __restrict__ avec, i64 sA,
    float* __restrict__ f1, float* __restrict__ f2, i64 sF) {
  int batch = blockIdx.y;
  Wh += batch * sWh; avec += batch * sA;
  int lane = threadIdx.x & 31, wave = threadIdx.x >> 5;
  int row = blockIdx.x * 8 + wave;
  float s1 = 0.f, s2 = 0.f;
  const float* wr = Wh + (i64)row * D;
  for (int c = lane; c < D; c += 32) { float w = wr[c]; s1 += w * avec[c]; s2 += w * avec[D + c]; }
#pragma unroll
  for (int off = 16; off; off >>= 1) { s1 += __shfl_xor(s1, off, 32); s2 += __shfl_xor(s2, off, 32); }
  if (lane == 0) { f1[batch * sF + row] = s1; f2[batch * sF + row] = s2; }
}

// ---------------------------------------------------------------------------
// Fused masked-softmax attention + (att @ Wh) via WMMA.
// Workgroup = 16 waves = 16 rows. LDS strip 16 x 4096 *f16* (row stride 4104
// halves: byte stride 8208 ≡ 16 mod 256 -> the 16 lanes' b128 row reads map to
// 16 distinct bank groups). ~131 KB LDS -> 2 workgroups per 320 KB WGP.
// Phase 1: e=leaky(f1[i]+f2[j]) masked -> f16; row max/sum in f32 regs;
//          strip <- exp(e-max) (f16, in [0,1]); rinv = 1/sum.
// Phase 2: wave w computes 16x16 column tile w of out = att @ Wh;
//          A frag = two ds b128 loads, B frag = 32B global from WhT[D,4096].
// ---------------------------------------------------------------------------
#define SSTRH 4104
__global__ __launch_bounds__(512) void attn_kernel(
    const float* __restrict__ adj,
    const float* __restrict__ f1, i64 sF1,
    const float* __restrict__ f2, i64 sF2,
    const _Float16* __restrict__ WhT, i64 sWhT,
    float* __restrict__ o32, i64 sO32, int ldo32,
    _Float16* __restrict__ o16, i64 sO16, int ldo16,
    int D, int act) {
  extern __shared__ char smemraw[];
  _Float16* strip = (_Float16*)smemraw;                 // [16][SSTRH]
  float*    rinv  = (float*)(smemraw + 16 * SSTRH * 2); // [16]
  int batch = blockIdx.y;
  f1 += batch * sF1; f2 += batch * sF2; WhT += batch * sWhT;
  int lane = threadIdx.x & 31, wave = threadIdx.x >> 5;
  int rowblk = blockIdx.x;
  int i = rowblk * 16 + wave;
  const float* arow = adj + (i64)i * NN;
  _Float16* srow = strip + wave * SSTRH;
  float fi = f1[i];
  float mx = -3.4e38f;
  for (int j = lane; j < NN; j += 32) {
    float e = leakyf(fi + f2[j]);
    e = arow[j] > 0.f ? e : NEGC;
    srow[j] = (_Float16)e;           // masked -> -inf in f16
    mx = fmaxf(mx, e);
  }
#pragma unroll
  for (int off = 16; off; off >>= 1) mx = fmaxf(mx, __shfl_xor(mx, off, 32));
  float sum = 0.f;
  for (int j = lane; j < NN; j += 32) {
    float w = __expf((float)srow[j] - mx);  // masked entries -> 0
    srow[j] = (_Float16)w;
    sum += w;
  }
#pragma unroll
  for (int off = 16; off; off >>= 1) sum += __shfl_xor(sum, off, 32);
  if (lane == 0) rinv[wave] = 1.0f / sum;
  __syncthreads();

  if (wave * 16 < D) {
    int hh = lane >> 4;
    int m = lane & 15;
    int ncol = wave * 16 + m;
    const _Float16* brow = WhT + (i64)ncol * NN;
    const _Float16* ar = strip + m * SSTRH;
    v8f acc = {};
    for (int kb = 0; kb < NN; kb += 32) {
      V16H a, b;
      a.q[0] = *(const uint4*)(ar + kb + hh * 16);
      a.q[1] = *(const uint4*)(ar + kb + hh * 16 + 8);
      b.q[0] = *(const uint4*)(brow + kb + hh * 16);
      b.q[1] = *(const uint4*)(brow + kb + hh * 16 + 8);
      acc = __builtin_amdgcn_wmma_f32_16x16x32_f16(false, a.v, false, b.v,
                                                   (short)0, acc, false, false);
    }
#pragma unroll
    for (int r = 0; r < 8; r++) {
      int mm = hh * 8 + r;
      float v = acc[r] * rinv[mm];
      if (act == 1) v = eluf(v);
      int irow = rowblk * 16 + mm;
      if (o32) o32[batch * sO32 + (i64)irow * ldo32 + ncol] = v;
      if (o16) o16[batch * sO16 + (i64)irow * ldo16 + ncol] = (_Float16)v;
    }
  }
}

// ---------------------------------------------------------------------------
// Whb = h @ Wo1 (D=64), f1b = Whb*ao1[0], f2b = Whb*ao1[1]
// ---------------------------------------------------------------------------
__global__ __launch_bounds__(256) void out1_prep_kernel(
    const float* __restrict__ h, const float* __restrict__ Wo1,
    const float* __restrict__ ao1, float* __restrict__ Whb,
    float* __restrict__ f1, float* __restrict__ f2) {
  int lane = threadIdx.x & 31, wave = threadIdx.x >> 5;
  int row = blockIdx.x * 8 + wave;
  float s = 0.f;
  const float* hr = h + (i64)row * 64;
  for (int c = lane; c < 64; c += 32) s += hr[c] * Wo1[c];
#pragma unroll
  for (int off = 16; off; off >>= 1) s += __shfl_xor(s, off, 32);
  if (lane == 0) { Whb[row] = s; f1[row] = s * ao1[0]; f2[row] = s * ao1[1]; }
}

// ---------------------------------------------------------------------------
// Final D=1 attention (two-pass, wave per row) + elu
// ---------------------------------------------------------------------------
__global__ __launch_bounds__(256) void attn_d1_kernel(
    const float* __restrict__ adj, const float* __restrict__ f1,
    const float* __restrict__ f2, const float* __restrict__ Whb,
    float* __restrict__ hout) {
  int lane = threadIdx.x & 31, wave = threadIdx.x >> 5;
  int i = blockIdx.x * 8 + wave;
  const float* arow = adj + (i64)i * NN;
  float fi = f1[i];
  float mx = -3.4e38f;
  for (int j = lane; j < NN; j += 32) {
    float e = leakyf(fi + f2[j]);
    e = arow[j] > 0.f ? e : NEGC;
    mx = fmaxf(mx, e);
  }
#pragma unroll
  for (int off = 16; off; off >>= 1) mx = fmaxf(mx, __shfl_xor(mx, off, 32));
  float s = 0.f, t = 0.f;
  for (int j = lane; j < NN; j += 32) {
    float e = leakyf(fi + f2[j]);
    e = arow[j] > 0.f ? e : NEGC;
    float w = __expf(e - mx);
    s += w; t += w * Whb[j];
  }
#pragma unroll
  for (int off = 16; off; off >>= 1) { s += __shfl_xor(s, off, 32); t += __shfl_xor(t, off, 32); }
  if (lane == 0) hout[i] = eluf(t / s);
}

// ---------------------------------------------------------------------------
// d_v = row sums of s_mat
// ---------------------------------------------------------------------------
__global__ __launch_bounds__(256) void rowsum_kernel(const float* __restrict__ s_mat,
                                                     float* __restrict__ dv) {
  int lane = threadIdx.x & 31, wave = threadIdx.x >> 5;
  int i = blockIdx.x * 8 + wave;
  float s = 0.f;
  const float* r = s_mat + (i64)i * NN;
  for (int j = lane; j < NN; j += 32) s += r[j];
#pragma unroll
  for (int off = 16; off; off >>= 1) s += __shfl_xor(s, off, 32);
  if (lane == 0) dv[i] = s;
}

// ---------------------------------------------------------------------------
// Degree MLP (thread per node)
// ---------------------------------------------------------------------------
__global__ void mlp_kernel(const float* __restrict__ h_, const float* __restrict__ dv,
                           const float* __restrict__ dW, const float* __restrict__ dW0,
                           const float* __restrict__ dW1, const float* __restrict__ dW01,
                           const float* __restrict__ dW2, const float* __restrict__ dW02,
                           const float* __restrict__ dV, const float* __restrict__ dV0,
                           float* __restrict__ out) {
  int i = blockIdx.x * blockDim.x + threadIdx.x;
  if (i >= NN) return;
  float x0 = h_[i], x1 = dv[i];
  float h0[10], h1[20], h2[10];
#pragma unroll
  for (int k = 0; k < 10; k++) h0[k] = leakyf(x0 * dW[k] + x1 * dW[10 + k] + dW0[k]);
#pragma unroll
  for (int k = 0; k < 20; k++) {
    float v = dW01[k];
    for (int c = 0; c < 10; c++) v += h0[c] * dW1[c * 20 + k];
    h1[k] = leakyf(v);
  }
#pragma unroll
  for (int k = 0; k < 10; k++) {
    float v = dW02[k];
    for (int c = 0; c < 20; c++) v += h1[c] * dW2[c * 10 + k];
    h2[k] = leakyf(v);
  }
  float o = dV0[0];
#pragma unroll
  for (int c = 0; c < 10; c++) o += h2[c] * dV[c];
  out[i] = leakyf(o);
}

// ---------------------------------------------------------------------------
// Host orchestration
// ---------------------------------------------------------------------------
static inline size_t alignup(size_t x) { return (x + 255) & ~(size_t)255; }

extern "C" void kernel_launch(void* const* d_in, const int* in_sizes, int n_in,
                              void* d_out, int out_size, void* d_ws, size_t ws_size,
                              hipStream_t stream) {
  const float* x      = (const float*)d_in[0];
  const float* adj    = (const float*)d_in[1];
  const int*   obs    = (const int*)  d_in[2];
  const float* s_mat  = (const float*)d_in[3];
  const float* theta  = (const float*)d_in[4];
  const float* Wh0    = (const float*)d_in[5];
  const float* ah0    = (const float*)d_in[6];
  const float* Wh1    = (const float*)d_in[7];
  const float* ah1    = (const float*)d_in[8];
  const float* Wo0    = (const float*)d_in[9];
  const float* ao0    = (const float*)d_in[10];
  const float* Wo1    = (const float*)d_in[11];
  const float* ao1    = (const float*)d_in[12];
  const float* dW     = (const float*)d_in[13];
  const float* dW0    = (const float*)d_in[14];
  const float* dW1    = (const float*)d_in[15];
  const float* dW01   = (const float*)d_in[16];
  const float* dW2    = (const float*)d_in[17];
  const float* dW02   = (const float*)d_in[18];
  const float* dV     = (const float*)d_in[19];
  const float* dV0    = (const float*)d_in[20];
  float* out = (float*)d_out;

  // workspace layout
  char* ws = (char*)d_ws;
  size_t off = 0;
  _Float16* hin   = (_Float16*)(ws + off); off = alignup(off + (size_t)NN * FF * 2);
  _Float16* W0T   = (_Float16*)(ws + off); off = alignup(off + (size_t)HH * DD1 * FF * 2);
  _Float16* W1T   = (_Float16*)(ws + off); off = alignup(off + (size_t)HH * DD1 * DD1 * 2);
  _Float16* WO0T  = (_Float16*)(ws + off); off = alignup(off + (size_t)OO1 * (HH * DD1) * 2);
  float*    Wh32  = (float*)   (ws + off); off = alignup(off + (size_t)HH * NN * DD1 * 4);
  _Float16* WhT16 = (_Float16*)(ws + off); off = alignup(off + (size_t)HH * DD1 * NN * 2);
  float*    f1    = (float*)   (ws + off); off = alignup(off + (size_t)HH * NN * 4);
  float*    f2    = (float*)   (ws + off); off = alignup(off + (size_t)HH * NN * 4);
  _Float16* h1f16 = (_Float16*)(ws + off); off = alignup(off + (size_t)HH * NN * DD1 * 2);
  _Float16* hcat  = (_Float16*)(ws + off); off = alignup(off + (size_t)NN * HH * DD1 * 2);
  float*    Who32 = (float*)   (ws + off); off = alignup(off + (size_t)NN * OO1 * 4);
  _Float16* WhoT  = (_Float16*)(ws + off); off = alignup(off + (size_t)OO1 * NN * 2);
  float*    hout0 = (float*)   (ws + off); off = alignup(off + (size_t)NN * OO1 * 4);
  float*    Whb   = (float*)   (ws + off); off = alignup(off + (size_t)NN * 4);
  float*    f1b   = (float*)   (ws + off); off = alignup(off + (size_t)NN * 4);
  float*    f2b   = (float*)   (ws + off); off = alignup(off + (size_t)NN * 4);
  float*    hfin  = (float*)   (ws + off); off = alignup(off + (size_t)NN * 4);
  float*    dv    = (float*)   (ws + off); off = alignup(off + (size_t)NN * 4);
  (void)ws_size; (void)n_in; (void)in_sizes; (void)out_size;

  const int smem_bytes = 16 * SSTRH * 2 + 64;
  (void)hipFuncSetAttribute((const void*)attn_kernel,
                            hipFuncAttributeMaxDynamicSharedMemorySize, smem_bytes);

  // 1) h_in (f16)
  prep_hin_kernel<<<dim3((NN * FF + 255) / 256), dim3(256), 0, stream>>>(x, obs, theta, hin);

  // 2) transpose-convert weights to f16
  cvtT_kernel<<<dim3((FF * DD1 + 255) / 256, 1, HH), dim3(256), 0, stream>>>(
      Wh0, (i64)FF * DD1, W0T, (i64)DD1 * FF, FF, DD1);
  cvtT_kernel<<<dim3((DD1 * DD1 + 255) / 256, 1, HH), dim3(256), 0, stream>>>(
      Wh1, (i64)DD1 * DD1, W1T, (i64)DD1 * DD1, DD1, DD1);
  cvtT_kernel<<<dim3((HH * DD1 * OO1 + 255) / 256, 1, 1), dim3(256), 0, stream>>>(
      Wo0, 0, WO0T, 0, HH * DD1, OO1);

  // 3) layer-0 GEMM: Wh = h_in @ W0 (per head)  -> f32 + transposed f16
  wmma_gemm_kernel<<<dim3(4, NN / 16, HH), dim3(128), 0, stream>>>(
      hin, 0, W0T, (i64)DD1 * FF,
      Wh32, (i64)NN * DD1, WhT16, (i64)DD1 * NN, NN, DD1, FF);

  // 4) layer-0 attention coefficients
  coef_kernel<<<dim3(NN / 8, HH), dim3(256), 0, stream>>>(
      Wh32, (i64)NN * DD1, DD1, ah0, (i64)2 * DD1, f1, f2, (i64)NN);

  // 5) layer-0 attention + att@Wh + elu -> h1 (f16)
  attn_kernel<<<dim3(NN / 16, HH), dim3(512), smem_bytes, stream>>>(
      adj, f1, (i64)NN, f2, (i64)NN, WhT16, (i64)DD1 * NN,
      nullptr, 0, 0, h1f16, (i64)NN * DD1, DD1, DD1, 1);

  // 6) layer-1 GEMM: Wh = h1 @ W1 (per head)
  wmma_gemm_kernel<<<dim3(4, NN / 16, HH), dim3(128), 0, stream>>>(
      h1f16, (i64)NN * DD1, W1T, (i64)DD1 * DD1,
      Wh32, (i64)NN * DD1, WhT16, (i64)DD1 * NN, NN, DD1, DD1);

  // 7) layer-1 coefficients
  coef_kernel<<<dim3(NN / 8, HH), dim3(256), 0, stream>>>(
      Wh32, (i64)NN * DD1, DD1, ah1, (i64)2 * DD1, f1, f2, (i64)NN);

  // 8) layer-1 attention + elu -> write directly into hcat[N, H*D1] at col head*256
  attn_kernel<<<dim3(NN / 16, HH), dim3(512), smem_bytes, stream>>>(
      adj, f1, (i64)NN, f2, (i64)NN, WhT16, (i64)DD1 * NN,
      nullptr, 0, 0, hcat, (i64)DD1, HH * DD1, DD1, 1);

  // 9) output GEMM: Who = hcat @ Wo0  [4096,1024]x[1024,64]
  wmma_gemm_kernel<<<dim3(1, NN / 16, 1), dim3(128), 0, stream>>>(
      hcat, 0, WO0T, 0, Who32, 0, WhoT, 0, NN, OO1, HH * DD1);

  // 10) output coefficients
  coef_kernel<<<dim3(NN / 8, 1), dim3(256), 0, stream>>>(
      Who32, 0, OO1, ao0, 0, f1, f2, 0);

  // 11) output attention (no elu) -> hout0 f32 [4096,64]
  attn_kernel<<<dim3(NN / 16, 1), dim3(512), smem_bytes, stream>>>(
      adj, f1, 0, f2, 0, WhoT, 0,
      hout0, 0, OO1, nullptr, 0, 0, OO1, 0);

  // 12) Whb = hout0 @ Wo1, scalar attention coeffs
  out1_prep_kernel<<<dim3(NN / 8), dim3(256), 0, stream>>>(hout0, Wo1, ao1, Whb, f1b, f2b);

  // 13) final D=1 attention + elu
  attn_d1_kernel<<<dim3(NN / 8), dim3(256), 0, stream>>>(adj, f1b, f2b, Whb, hfin);

  // 14) degrees
  rowsum_kernel<<<dim3(NN / 8), dim3(256), 0, stream>>>(s_mat, dv);

  // 15) degree MLP -> output [4096,1]
  mlp_kernel<<<dim3(NN / 256), dim3(256), 0, stream>>>(
      hfin, dv, dW, dW0, dW1, dW01, dW2, dW02, dV, dV0, out);
}
